// DecoderPromptLayerWithNMR_79989470920967
// MI455X (gfx1250) — compile-verified, matrix operand order checked by reading, not run on hardware
//
#include <hip/hip_runtime.h>

// ---------------- problem constants ----------------
#define D_    1024
#define H_    16
#define DH_   64
#define ROT_  32
#define FFN_  4096
#define B_    4
#define T_    1024
#define MEM_  128
#define NMR_  64
#define PREF_ 192          // MEM_ + NMR_
#define L_    1216         // PREF_ + T_
#define SCALE_ 0.125f      // 1/sqrt(DH)

typedef __attribute__((ext_vector_type(16))) __bf16 bf16x16;
typedef __attribute__((ext_vector_type(8)))  __bf16 bf16x8;
typedef __attribute__((ext_vector_type(8)))  float  f32x8;

// Build a 16x16x32 bf16 WMMA A/B fragment from memory laid out with the K-run
// contiguous: elems 0..7 = p[0..7] (K = khalf+0..7), elems 8..15 = p[16..23]
// (K = khalf+16..23). Caller passes p = row_base + hi*8.
__device__ __forceinline__ bf16x16 frag16(const __bf16* p) {
  union { bf16x16 v; bf16x8 h[2]; } u;
  u.h[0] = *(const bf16x8*)(p);
  u.h[1] = *(const bf16x8*)(p + 16);
  return u.v;
}

// ---- CDNA5 async global->LDS copy (ASYNCcnt path), 16B per lane ----
// VDST = per-lane LDS byte address (low 32 bits of the generic LDS pointer),
// VADDR = 64-bit global address, no SADDR.  See CDNA5 ISA.
__device__ __forceinline__ void async_copy_b128(const void* gsrc, void* ldst) {
  unsigned lds_addr = (unsigned)(size_t)ldst;           // LDS offset in [31:0]
  unsigned long long gaddr = (unsigned long long)(size_t)gsrc;
  asm volatile("global_load_async_to_lds_b128 %0, %1, off"
               :: "v"(lds_addr), "v"(gaddr)
               : "memory");
}
__device__ __forceinline__ void wait_async0() {
  asm volatile("s_wait_asynccnt 0" ::: "memory");
}

// ---------------- weight transpose + fp32->bf16 ----------------
// W is (K x Ncols) row-major fp32; Wt becomes (Ncols x K) row-major bf16.
__global__ __launch_bounds__(256, 1)
void transpose_bf16_kernel(const float* __restrict__ W,
                           __bf16* __restrict__ Wt,
                           int K, int Ncols) {
  __shared__ float tile[32][33];
  int k0 = blockIdx.x * 32;
  int n0 = blockIdx.y * 32;
  int tx = threadIdx.x, ty = threadIdx.y;   // 32 x 8
#pragma unroll
  for (int i = 0; i < 32; i += 8)
    tile[ty + i][tx] = W[(size_t)(k0 + ty + i) * Ncols + n0 + tx];
  __syncthreads();
#pragma unroll
  for (int i = 0; i < 32; i += 8)
    Wt[(size_t)(n0 + ty + i) * K + k0 + tx] = (__bf16)tile[tx][ty + i];
}

// ---------------- LayerNorm kernels ----------------
__device__ __forceinline__ void block_stats(float s, float sq, float* sb,
                                            float& mu, float& rs) {
#pragma unroll
  for (int m = 16; m; m >>= 1) { s += __shfl_xor(s, m, 32); sq += __shfl_xor(sq, m, 32); }
  int wv = threadIdx.x >> 5, ln = threadIdx.x & 31;
  if (ln == 0) { sb[wv] = s; sb[8 + wv] = sq; }
  __syncthreads();
  if (threadIdx.x < 32) {
    float a = (ln < 8) ? sb[ln] : 0.f;
    float b = (ln < 8) ? sb[8 + ln] : 0.f;
#pragma unroll
    for (int m = 4; m; m >>= 1) { a += __shfl_xor(a, m, 32); b += __shfl_xor(b, m, 32); }
    if (ln == 0) { sb[0] = a; sb[8] = b; }
  }
  __syncthreads();
  mu = sb[0] * (1.f / D_);
  float var = sb[8] * (1.f / D_) - mu * mu;
  rs = rsqrtf(var + 1e-5f);
}

// concat(memory, nmr, x) row -> xfull fp32 + LN1 -> xa bf16
__global__ __launch_bounds__(256, 1)
void ln1_concat_kernel(const float* __restrict__ x,
                       const float* __restrict__ mem,
                       const float* __restrict__ nmr,
                       const float* __restrict__ g,
                       const float* __restrict__ beta,
                       float* __restrict__ xfull,
                       __bf16* __restrict__ xa) {
  int l = blockIdx.x, b = blockIdx.y;
  const float* src;
  if (l < MEM_)       src = mem + ((size_t)b * MEM_ + l) * D_;
  else if (l < PREF_) src = nmr + ((size_t)b * NMR_ + (l - MEM_)) * D_;
  else                src = x   + ((size_t)b * T_   + (l - PREF_)) * D_;
  int tid = threadIdx.x;
  float v[4], s = 0.f, sq = 0.f;
#pragma unroll
  for (int i = 0; i < 4; ++i) { v[i] = src[tid + i * 256]; s += v[i]; sq += v[i] * v[i]; }
  __shared__ float sb[16];
  float mu, rs;
  block_stats(s, sq, sb, mu, rs);
#pragma unroll
  for (int i = 0; i < 4; ++i) {
    int c = tid + i * 256;
    size_t o = ((size_t)b * L_ + l) * D_ + c;
    xfull[o] = v[i];
    xa[o]    = (__bf16)((v[i] - mu) * rs * g[c] + beta[c]);
  }
}

// LN2: y fp32 row -> yf bf16
__global__ __launch_bounds__(256, 1)
void ln2_kernel(const float* __restrict__ y,
                const float* __restrict__ g,
                const float* __restrict__ beta,
                __bf16* __restrict__ yf) {
  size_t row = blockIdx.x;
  const float* src = y + row * D_;
  int tid = threadIdx.x;
  float v[4], s = 0.f, sq = 0.f;
#pragma unroll
  for (int i = 0; i < 4; ++i) { v[i] = src[tid + i * 256]; s += v[i]; sq += v[i] * v[i]; }
  __shared__ float sb[16];
  float mu, rs;
  block_stats(s, sq, sb, mu, rs);
#pragma unroll
  for (int i = 0; i < 4; ++i) {
    int c = tid + i * 256;
    yf[row * D_ + c] = (__bf16)((v[i] - mu) * rs * g[c] + beta[c]);
  }
}

// ---------------- generic 64x64 WMMA GEMM ----------------
// Double-buffered LDS staging via CDNA5 async global->LDS copies: the WMMAs of
// step k overlap the async fetch of step k+1; sync = s_wait_asynccnt 0 + barrier.
// C = A(Mrows x K, bf16 row-major) * Bt^T (Bt is Ncols x K bf16 row-major) + bias
// Epilogues:
//  EPI 0: Q/K proj: +bias, RoPE on cols [0,32) of the head, write bf16 (B,H,L,DH)
//  EPI 1: V proj:   +bias, write bf16 (B,H,DH,L) (transposed for P*V B-fragments)
//  EPI 2: +bias +resid -> fp32 row-major.  residMode1: resid = xfull[(b*L+PREF+tg)*D+c]
//  EPI 3: relu(acc+bias)^2 -> bf16 row-major
#define LDT 40   // LDS row stride (elems): 80B rows -> conflict-free frag reads
template <int EPI>
__global__ __launch_bounds__(128, 1)
void gemm64_kernel(const __bf16* __restrict__ A,
                   const __bf16* __restrict__ Bt,
                   const float* __restrict__ bias,
                   const float* __restrict__ resid,
                   void* __restrict__ outp,
                   int Mrows, int K, int Ncols, int residMode) {
  __shared__ __align__(16) __bf16 As[2][64][LDT];
  __shared__ __align__(16) __bf16 Bs[2][64][LDT];
  const int row0 = blockIdx.x * 64;
  const int n0   = blockIdx.y * 64;
  const int tid  = threadIdx.x;            // 128 threads = 4 waves
  const int w    = tid >> 5;
  const int lane = tid & 31;
  const int lh   = lane & 15;
  const int hi   = lane >> 4;

  // Per-thread cooperative-copy coordinates (2 x 16B for A and B each step)
  int ldr[2], lds_[2];
#pragma unroll
  for (int t2 = 0; t2 < 2; ++t2) {
    int idx = tid + t2 * 128;
    ldr[t2] = idx >> 2;        // tile row
    lds_[t2] = (idx & 3) * 8;  // k-offset within 32
  }

  f32x8 acc[4];
#pragma unroll
  for (int nt = 0; nt < 4; ++nt)
#pragma unroll
    for (int e = 0; e < 8; ++e) acc[nt][e] = 0.f;

  // Prologue: async-stage k-step 0 into buffer 0
#pragma unroll
  for (int t2 = 0; t2 < 2; ++t2) {
    async_copy_b128(&A[(size_t)(row0 + ldr[t2]) * K + lds_[t2]],
                    &As[0][ldr[t2]][lds_[t2]]);
    async_copy_b128(&Bt[(size_t)(n0 + ldr[t2]) * K + lds_[t2]],
                    &Bs[0][ldr[t2]][lds_[t2]]);
  }
  wait_async0();
  __syncthreads();

  const int nsteps = K >> 5;
  int buf = 0;
  for (int step = 0; step < nsteps; ++step) {
    const bool more = (step + 1 < nsteps);
    if (more) {  // launch next tile's async copies before this step's WMMAs
      int kk = (step + 1) << 5;
#pragma unroll
      for (int t2 = 0; t2 < 2; ++t2) {
        async_copy_b128(&A[(size_t)(row0 + ldr[t2]) * K + kk + lds_[t2]],
                        &As[buf ^ 1][ldr[t2]][lds_[t2]]);
        async_copy_b128(&Bt[(size_t)(n0 + ldr[t2]) * K + kk + lds_[t2]],
                        &Bs[buf ^ 1][ldr[t2]][lds_[t2]]);
      }
    }
    bf16x16 af = frag16(&As[buf][w * 16 + lh][hi * 8]);
#pragma unroll
    for (int nt = 0; nt < 4; ++nt) {
      bf16x16 bfr = frag16(&Bs[buf][nt * 16 + lh][hi * 8]);
      acc[nt] = __builtin_amdgcn_wmma_f32_16x16x32_bf16(
          false, af, false, bfr, (short)0, acc[nt], false, false);
    }
    if (more) {
      wait_async0();
      __syncthreads();
      buf ^= 1;
    }
  }

  // bias
#pragma unroll
  for (int nt = 0; nt < 4; ++nt) {
    float bv = bias[n0 + nt * 16 + lh];
#pragma unroll
    for (int e = 0; e < 8; ++e) acc[nt][e] += bv;
  }

  if (EPI == 0) {  // RoPE on first 32 cols of the head (acc0 <-> acc1 pair up)
    float invf = __powf(10000.f, -((float)lh) * (1.f / 16.f));
#pragma unroll
    for (int e = 0; e < 8; ++e) {
      int r = row0 + w * 16 + e + 8 * hi;
      int l = r % L_;
      float ang = (float)l * invf;
      float cs = __cosf(ang), sn = __sinf(ang);
      float v0 = acc[0][e], v1 = acc[1][e];
      acc[0][e] = v0 * cs - v1 * sn;
      acc[1][e] = v1 * cs + v0 * sn;
    }
  }

  if (EPI == 0 || EPI == 1) {
    __bf16* out = (__bf16*)outp;
    int h = n0 >> 6;
#pragma unroll
    for (int nt = 0; nt < 4; ++nt) {
      int c = nt * 16 + lh;
#pragma unroll
      for (int e = 0; e < 8; ++e) {
        int r = row0 + w * 16 + e + 8 * hi;
        int b = r / L_, l = r - b * L_;
        size_t o = (EPI == 0)
            ? (((size_t)(b * H_ + h) * L_ + l) * DH_ + c)     // (B,H,L,DH)
            : (((size_t)(b * H_ + h) * DH_ + c) * L_ + l);    // (B,H,DH,L)
        out[o] = (__bf16)acc[nt][e];
      }
    }
  } else if (EPI == 2) {
    float* out = (float*)outp;
#pragma unroll
    for (int nt = 0; nt < 4; ++nt) {
      int c = n0 + nt * 16 + lh;
#pragma unroll
      for (int e = 0; e < 8; ++e) {
        int r = row0 + w * 16 + e + 8 * hi;
        size_t ri;
        if (residMode) { int b = r >> 10, tg = r & 1023;
          ri = ((size_t)(b * L_ + PREF_ + tg)) * D_ + c; }
        else ri = (size_t)r * Ncols + c;
        out[(size_t)r * Ncols + c] = acc[nt][e] + resid[ri];
      }
    }
  } else {  // EPI 3: squared ReLU -> bf16
    __bf16* out = (__bf16*)outp;
#pragma unroll
    for (int nt = 0; nt < 4; ++nt) {
      int c = n0 + nt * 16 + lh;
#pragma unroll
      for (int e = 0; e < 8; ++e) {
        int r = row0 + w * 16 + e + 8 * hi;
        float v = fmaxf(acc[nt][e], 0.f);
        out[(size_t)r * Ncols + c] = (__bf16)(v * v);
      }
    }
  }
}

// ---------------- flash attention ----------------
// grid (T/64, H, B), 128 threads (4 waves x 16 q-rows). Prefix keys always
// visible; causal among suffix; only suffix queries computed.
__global__ __launch_bounds__(128, 1)
void flash_attn_kernel(const __bf16* __restrict__ Qh,
                       const __bf16* __restrict__ Kh,
                       const __bf16* __restrict__ Vt,
                       __bf16* __restrict__ attn) {
  const int qt = blockIdx.x, h = blockIdx.y, b = blockIdx.z;
  const int w = threadIdx.x >> 5, lane = threadIdx.x & 31;
  const int lh = lane & 15, hi = lane >> 4;

  const __bf16* Qb = Qh + ((size_t)(b * H_ + h) * L_) * DH_;
  const __bf16* Kb = Kh + ((size_t)(b * H_ + h) * L_) * DH_;
  const __bf16* Vb = Vt + ((size_t)(b * H_ + h) * DH_) * L_;

  __shared__ __align__(16) __bf16 Pl[4][16][72];  // wave-private P transpose

  // Q fragments held in registers for the whole tile
  int qr = PREF_ + qt * 64 + w * 16 + lh;
  bf16x16 aq[2];
#pragma unroll
  for (int ks = 0; ks < 2; ++ks)
    aq[ks] = frag16(Qb + (size_t)qr * DH_ + ks * 32 + hi * 8);

  f32x8 o[4];
  float mrun[8], lrun[8];
#pragma unroll
  for (int nt = 0; nt < 4; ++nt)
#pragma unroll
    for (int e = 0; e < 8; ++e) o[nt][e] = 0.f;
#pragma unroll
  for (int e = 0; e < 8; ++e) { mrun[e] = -1e30f; lrun[e] = 0.f; }

  const int ktmax = 3 + qt;  // 3 prefix tiles + causal tiles
  for (int kt = 0; kt <= ktmax; ++kt) {
    // prefetch next key/value tile toward the WGP (global_prefetch_b8)
    if (kt < ktmax) {
      __builtin_prefetch(Kb + (size_t)((kt + 1) * 64 + lane * 2) * DH_, 0, 1);
      __builtin_prefetch(Vb + (size_t)(lane * 2) * L_ + (kt + 1) * 64, 0, 1);
    }
    // ---- S = Q K^T for this 64-key tile ----
    f32x8 s[4];
#pragma unroll
    for (int nt = 0; nt < 4; ++nt) {
#pragma unroll
      for (int e = 0; e < 8; ++e) s[nt][e] = 0.f;
      int key = kt * 64 + nt * 16 + lh;
#pragma unroll
      for (int ks = 0; ks < 2; ++ks) {
        bf16x16 bk = frag16(Kb + (size_t)key * DH_ + ks * 32 + hi * 8);
        s[nt] = __builtin_amdgcn_wmma_f32_16x16x32_bf16(
            false, aq[ks], false, bk, (short)0, s[nt], false, false);
      }
    }
    // ---- scale + diagonal causal mask (in place on s) ----
    const bool diag = (kt == ktmax);
#pragma unroll
    for (int nt = 0; nt < 4; ++nt) {
      int j = kt * 64 + nt * 16 + lh;
#pragma unroll
      for (int e = 0; e < 8; ++e) {
        float sv = s[nt][e] * SCALE_;
        if (diag && j > (PREF_ + qt * 64 + w * 16 + e + 8 * hi)) sv = -1e30f;
        s[nt][e] = sv;
      }
    }
    // ---- online softmax (row spread over 16 lanes x 4 accums), in place ----
#pragma unroll
    for (int e = 0; e < 8; ++e) {
      float rm = fmaxf(fmaxf(s[0][e], s[1][e]), fmaxf(s[2][e], s[3][e]));
#pragma unroll
      for (int m = 1; m < 16; m <<= 1) rm = fmaxf(rm, __shfl_xor(rm, m, 32));
      float mnew = fmaxf(mrun[e], rm);
      float alpha = __expf(mrun[e] - mnew);
      float rsum = 0.f;
#pragma unroll
      for (int nt = 0; nt < 4; ++nt) {
        float p = __expf(s[nt][e] - mnew);
        s[nt][e] = p;
        rsum += p;
      }
#pragma unroll
      for (int m = 1; m < 16; m <<= 1) rsum += __shfl_xor(rsum, m, 32);
      lrun[e] = lrun[e] * alpha + rsum;
      mrun[e] = mnew;
#pragma unroll
      for (int nt = 0; nt < 4; ++nt) o[nt][e] *= alpha;
    }
    // ---- transpose P through LDS into A-fragment layout ----
    __syncthreads();
#pragma unroll
    for (int nt = 0; nt < 4; ++nt)
#pragma unroll
      for (int e = 0; e < 8; ++e)
        Pl[w][e + 8 * hi][nt * 16 + lh] = (__bf16)s[nt][e];
    __syncthreads();
    bf16x16 ap[2];
#pragma unroll
    for (int ks = 0; ks < 2; ++ks)
      ap[ks] = frag16(&Pl[w][lh][ks * 32 + hi * 8]);
    // ---- O += P V ----
#pragma unroll
    for (int nt = 0; nt < 4; ++nt) {
      int dh = nt * 16 + lh;
#pragma unroll
      for (int ks = 0; ks < 2; ++ks) {
        bf16x16 bv = frag16(Vb + (size_t)dh * L_ + kt * 64 + ks * 32 + hi * 8);
        o[nt] = __builtin_amdgcn_wmma_f32_16x16x32_bf16(
            false, ap[ks], false, bv, (short)0, o[nt], false, false);
      }
    }
  }

  // ---- normalize + store attn (B, T, D) bf16 ----
#pragma unroll
  for (int nt = 0; nt < 4; ++nt) {
    int c = h * 64 + nt * 16 + lh;
#pragma unroll
    for (int e = 0; e < 8; ++e) {
      int tg = qt * 64 + w * 16 + e + 8 * hi;
      attn[((size_t)b * T_ + tg) * D_ + c] = (__bf16)(o[nt][e] / lrun[e]);
    }
  }
}

// ---------------- launcher ----------------
extern "C" void kernel_launch(void* const* d_in, const int* in_sizes, int n_in,
                              void* d_out, int out_size, void* d_ws, size_t ws_size,
                              hipStream_t stream) {
  const float* x    = (const float*)d_in[0];
  const float* mem  = (const float*)d_in[1];
  const float* nmr  = (const float*)d_in[2];
  const float* Wq   = (const float*)d_in[3];
  const float* bq   = (const float*)d_in[4];
  const float* Wk   = (const float*)d_in[5];
  const float* bk   = (const float*)d_in[6];
  const float* Wv   = (const float*)d_in[7];
  const float* bv   = (const float*)d_in[8];
  const float* Wo   = (const float*)d_in[9];
  const float* bo   = (const float*)d_in[10];
  const float* W1   = (const float*)d_in[11];
  const float* b1   = (const float*)d_in[12];
  const float* W2   = (const float*)d_in[13];
  const float* b2   = (const float*)d_in[14];
  const float* ln1g = (const float*)d_in[15];
  const float* ln1b = (const float*)d_in[16];
  const float* ln2g = (const float*)d_in[17];
  const float* ln2b = (const float*)d_in[18];
  float* out = (float*)d_out;

  char* wp = (char*)d_ws;
  auto alloc = [&](size_t bytes) {
    void* p = (void*)wp;
    wp += (bytes + 255) & ~(size_t)255;
    return p;
  };
  float*  xfull = (float*) alloc((size_t)B_ * L_ * D_ * 4);
  __bf16* xa    = (__bf16*)alloc((size_t)B_ * L_ * D_ * 2);
  __bf16* Wqt   = (__bf16*)alloc((size_t)D_ * D_ * 2);
  __bf16* Wkt   = (__bf16*)alloc((size_t)D_ * D_ * 2);
  __bf16* Wvt   = (__bf16*)alloc((size_t)D_ * D_ * 2);
  __bf16* Wot   = (__bf16*)alloc((size_t)D_ * D_ * 2);
  __bf16* W1t   = (__bf16*)alloc((size_t)D_ * FFN_ * 2);
  __bf16* W2t   = (__bf16*)alloc((size_t)D_ * FFN_ * 2);
  __bf16* Qh    = (__bf16*)alloc((size_t)B_ * L_ * D_ * 2);
  __bf16* Kh    = (__bf16*)alloc((size_t)B_ * L_ * D_ * 2);
  __bf16* Vt2   = (__bf16*)alloc((size_t)B_ * L_ * D_ * 2);
  __bf16* attn  = (__bf16*)alloc((size_t)B_ * T_ * D_ * 2);
  float*  y     = (float*) alloc((size_t)B_ * T_ * D_ * 4);
  __bf16* yf    = (__bf16*)alloc((size_t)B_ * T_ * D_ * 2);
  __bf16* hmid  = (__bf16*)alloc((size_t)B_ * T_ * FFN_ * 2);

  dim3 tb(32, 8);
  transpose_bf16_kernel<<<dim3(D_/32,  D_/32 ), tb, 0, stream>>>(Wq, Wqt, D_,  D_);
  transpose_bf16_kernel<<<dim3(D_/32,  D_/32 ), tb, 0, stream>>>(Wk, Wkt, D_,  D_);
  transpose_bf16_kernel<<<dim3(D_/32,  D_/32 ), tb, 0, stream>>>(Wv, Wvt, D_,  D_);
  transpose_bf16_kernel<<<dim3(D_/32,  D_/32 ), tb, 0, stream>>>(Wo, Wot, D_,  D_);
  transpose_bf16_kernel<<<dim3(D_/32,  FFN_/32), tb, 0, stream>>>(W1, W1t, D_,  FFN_);
  transpose_bf16_kernel<<<dim3(FFN_/32, D_/32 ), tb, 0, stream>>>(W2, W2t, FFN_, D_);

  ln1_concat_kernel<<<dim3(L_, B_), 256, 0, stream>>>(x, mem, nmr, ln1g, ln1b,
                                                      xfull, xa);

  const int MQKV = B_ * L_;  // 4864 rows, 76 blocks
  gemm64_kernel<0><<<dim3(MQKV/64, D_/64), 128, 0, stream>>>(
      xa, Wqt, bq, nullptr, Qh, MQKV, D_, D_, 0);
  gemm64_kernel<0><<<dim3(MQKV/64, D_/64), 128, 0, stream>>>(
      xa, Wkt, bk, nullptr, Kh, MQKV, D_, D_, 0);
  gemm64_kernel<1><<<dim3(MQKV/64, D_/64), 128, 0, stream>>>(
      xa, Wvt, bv, nullptr, Vt2, MQKV, D_, D_, 0);

  flash_attn_kernel<<<dim3(T_/64, H_, B_), 128, 0, stream>>>(Qh, Kh, Vt2, attn);

  const int MT = B_ * T_;    // 4096 rows
  gemm64_kernel<2><<<dim3(MT/64, D_/64), 128, 0, stream>>>(
      attn, Wot, bo, xfull, y, MT, D_, D_, 1);

  ln2_kernel<<<dim3(MT), 256, 0, stream>>>(y, ln2g, ln2b, yf);

  gemm64_kernel<3><<<dim3(MT/64, FFN_/64), 128, 0, stream>>>(
      yf, W1t, b1, nullptr, hmid, MT, D_, FFN_, 0);

  gemm64_kernel<2><<<dim3(MT/64, D_/64), 128, 0, stream>>>(
      hmid, W2t, b2, y, out, MT, FFN_, D_, 0);
}
